// GraphDistillOperatorWithEdgeWeight_28613072126557
// MI455X (gfx1250) — compile-verified, matrix operand order checked by reading, not run on hardware
//
#include <hip/hip_runtime.h>
#include <hip/hip_bf16.h>
#include <stdint.h>

#define NROW 8192
#define DFEAT 256
#define NOUT 256
#define LDS_STRIDE 40  // 40 bf16 = 80B row stride -> conflict-free b128 LDS reads
#define NKSTEP (NROW / 32)

typedef __attribute__((ext_vector_type(16))) __bf16 v16bf;
typedef __attribute__((ext_vector_type(8)))  __bf16 bf16x8;
typedef __attribute__((ext_vector_type(8)))  float  v8f;
typedef __attribute__((ext_vector_type(4)))  float  f32x4;

#if __has_builtin(__builtin_amdgcn_global_load_async_to_lds_b128)
#define USE_ASYNC_LDS 1
#else
#define USE_ASYNC_LDS 0
#endif

#if USE_ASYNC_LDS
// Builtin expects pointers to 16-byte int vectors: global (AS1) src, LDS (AS3) dst.
typedef int i32x4 __attribute__((vector_size(4 * sizeof(int))));
typedef __attribute__((address_space(1))) i32x4 g_i32x4;
typedef __attribute__((address_space(3))) i32x4 l_i32x4;

static __device__ __forceinline__ g_i32x4* to_global(const void* p) {
  return (g_i32x4*)(uintptr_t)p;
}
// Generic pointers to __shared__ carry the LDS byte offset in their low 32 bits
// (per the ISA aperture rule), so truncation yields the AS3 ptr.
static __device__ __forceinline__ l_i32x4* to_lds(void* p) {
  return (l_i32x4*)(uint32_t)(uintptr_t)p;
}

static __device__ __forceinline__ void wait_async0() {
#if __has_builtin(__builtin_amdgcn_s_wait_asynccnt)
  __builtin_amdgcn_s_wait_asynccnt(0);
#else
  asm volatile("s_wait_asynccnt 0x0" ::: "memory");
#endif
}
#endif  // USE_ASYNC_LDS

static __device__ __forceinline__ float fast_exp2(float x) {
#if __has_builtin(__builtin_amdgcn_exp2f)
  return __builtin_amdgcn_exp2f(x);
#else
  return exp2f(x);
#endif
}

static __device__ __forceinline__ float fast_rcp(float x) {
#if __has_builtin(__builtin_amdgcn_rcpf)
  return __builtin_amdgcn_rcpf(x);
#else
  return 1.0f / x;
#endif
}

static __device__ __forceinline__ v8f zero_v8f() {
  v8f z;
#pragma unroll
  for (int i = 0; i < 8; ++i) z[i] = 0.0f;
  return z;
}

// Assemble a 16-element bf16 WMMA fragment from two 8-element (128-bit) loads.
// Per-lane layout (16-bit A and B): elems 0..7 = K(hi*8 + 0..7),
// elems 8..15 = K(hi*8 + 16..23); caller passes pointers at +0 and +16 elems.
static __device__ __forceinline__ v16bf make_frag(const __bf16* lo, const __bf16* hi) {
  bf16x8 l = *(const bf16x8*)lo;
  bf16x8 h = *(const bf16x8*)hi;
  v16bf r;
#pragma unroll
  for (int i = 0; i < 8; ++i) { r[i] = l[i]; r[i + 8] = h[i]; }
  return r;
}

// ---------------- prep: bf16 features (row-major + transposed) ----------------
__global__ void k_prep_feats(const float* __restrict__ f,
                             __bf16* __restrict__ fbf,
                             __bf16* __restrict__ fT) {
  int idx = blockIdx.x * blockDim.x + threadIdx.x;
  if (idx >= NROW * DFEAT) return;
  int i = idx >> 8;     // row   (D = 256)
  int d = idx & 255;    // col
  __bf16 h = (__bf16)f[idx];
  fbf[idx] = h;
  fT[(size_t)d * NROW + i] = h;
}

// ---- prep: combined weight B[k,o] = (k<256 ? W_out[o,k] : 0) - W_distill[o,k]
// stored as Bc[o][k] (row-major [256,512]) so B-fragments read K-contiguous. ----
__global__ void k_prep_w(const float* __restrict__ Wd,
                         const float* __restrict__ Wo,
                         __bf16* __restrict__ Bc) {
  int idx = blockIdx.x * blockDim.x + threadIdx.x;
  if (idx >= 256 * 512) return;
  int o = idx >> 9;
  int k = idx & 511;
  float v = ((k < 256) ? Wo[o * 256 + k] : 0.0f) - Wd[idx];
  Bc[idx] = (__bf16)v;
}

// ---------------- fused masked-softmax @ features (flash style) ----------------
// One pass over adj (256 MB): e = exp2(5*log2e * adj); tail = (e @ f) / rowsum(e)
// with exact diagonal (self-loop) correction. 4 waves/WG share a double-buffered
// LDS-staged 32x256 bf16 B tile of features^T, filled with async global->LDS DMA
// (ASYNCcnt) one iteration ahead. adj A-tiles are register-prefetched one
// iteration ahead; the exp/cvt A-fragment build runs before the barrier, in the
// shadow of the cross-wave sync. B fragments are software-pipelined one tile
// ahead so each WMMA overlaps the next tile's ds_load_b128 pair.
__global__ void __launch_bounds__(128)
k_softmax_spmm(const float* __restrict__ adj, const __bf16* __restrict__ fT,
               const __bf16* __restrict__ fbf, __bf16* __restrict__ tailbf) {
  __shared__ __bf16 tileB[2][DFEAT * LDS_STRIDE];  // 2 x 20 KB

  const int tid  = threadIdx.x;
  const int lane = tid & 31;
  const int wave = tid >> 5;
  const int ln   = lane & 15;
  const int hi   = lane >> 4;
  const int i0   = (blockIdx.x * 4 + wave) * 16;   // 16-row strip per wave

  v8f acc[16];
#pragma unroll
  for (int t = 0; t < 16; ++t) acc[t] = zero_v8f();
  v8f sacc = zero_v8f();

  v16bf ones;
#pragma unroll
  for (int i = 0; i < 16; ++i) ones[i] = (__bf16)1.0f;

  const float C5 = 7.2134752044448169f;  // 5 * log2(e)
  const size_t arow = (size_t)(i0 + ln) * NROW;
  const float* aprow = adj + arow + hi * 8;

  // Stage fT[:, j0..j0+31] into tileB[buf]; 2 rows per thread, 4 x b128 per row.
  auto stage = [&](int buf, int j0) {
#pragma unroll
    for (int r = 0; r < 2; ++r) {
      const int d = tid + r * 128;
      const __bf16* src = fT + (size_t)d * NROW + j0;
      __bf16* dst = &tileB[buf][d * LDS_STRIDE];
#pragma unroll
      for (int c = 0; c < 4; ++c) {
#if USE_ASYNC_LDS
        __builtin_amdgcn_global_load_async_to_lds_b128(
            to_global(src + c * 8), to_lds(dst + c * 8), 0, 0);
#else
        *(bf16x8*)(dst + c * 8) = *(const bf16x8*)(src + c * 8);
#endif
      }
    }
  };

  // Prologue: stage first B tile, prefetch first A tile.
  stage(0, 0);
  f32x4 x0 = *(const f32x4*)(aprow);
  f32x4 x1 = *(const f32x4*)(aprow + 4);
  f32x4 x2 = *(const f32x4*)(aprow + 16);
  f32x4 x3 = *(const f32x4*)(aprow + 20);

  for (int jb = 0; jb < NKSTEP; ++jb) {
    const int cur = jb & 1;

    // Build A fragment (e = exp(5*adj) in bf16) BEFORE the sync: three
    // independent passes (mul / exp / cvt) so TRANS latency self-hides, and the
    // whole block executes in the shadow of the barrier wait.
    float ev[16];
#pragma unroll
    for (int i = 0; i < 4; ++i) {
      ev[i]      = x0[i] * C5;
      ev[4 + i]  = x1[i] * C5;
      ev[8 + i]  = x2[i] * C5;
      ev[12 + i] = x3[i] * C5;
    }
#pragma unroll
    for (int i = 0; i < 16; ++i) ev[i] = fast_exp2(ev[i]);
    v16bf va;
#pragma unroll
    for (int i = 0; i < 16; ++i) va[i] = (__bf16)ev[i];

#if USE_ASYNC_LDS
    wait_async0();        // my async transfers for `cur` have landed in LDS
#endif
    __syncthreads();      // everyone's transfers landed; `cur^1` is free to refill

    if (jb + 1 < NKSTEP) stage(cur ^ 1, (jb + 1) * 32);

    // Prefetch next adj A tile (latency hidden behind this step's WMMAs).
    f32x4 n0 = x0, n1 = x1, n2 = x2, n3 = x3;
    if (jb + 1 < NKSTEP) {
      const float* ap = aprow + (jb + 1) * 32;
      n0 = *(const f32x4*)(ap);
      n1 = *(const f32x4*)(ap + 4);
      n2 = *(const f32x4*)(ap + 16);
      n3 = *(const f32x4*)(ap + 20);
    }

    // Softmax denominator via ones-B (every N column holds the rowsum).
    sacc = __builtin_amdgcn_wmma_f32_16x16x32_bf16(false, va, false, ones,
                                                   (short)0, sacc, false, false);

    // B fragments software-pipelined one tile ahead: ds_loads for t+1 are in
    // flight while WMMA t executes.
    const __bf16* pbase = &tileB[cur][ln * LDS_STRIDE + hi * 8];
    v16bf vb = make_frag(pbase, pbase + 16);
#pragma unroll
    for (int t = 0; t < 16; ++t) {
      v16bf vbn = vb;
      if (t + 1 < 16) {
        const __bf16* p = pbase + (t + 1) * 16 * LDS_STRIDE;
        vbn = make_frag(p, p + 16);
      }
      acc[t] = __builtin_amdgcn_wmma_f32_16x16x32_bf16(false, va, false, vb,
                                                       (short)0, acc[t], false, false);
      vb = vbn;
    }

    x0 = n0; x1 = n1; x2 = n2; x3 = n3;
  }

  // Diagonal (self-loop mask) correction + normalize + store bf16 tail.
  float ediag = fast_exp2(C5 * adj[(size_t)(i0 + ln) * NROW + (i0 + ln)]);
  ediag = (float)(__bf16)ediag;  // match bf16 quantization used in accumulation
  float esel[8], inv[8];
#pragma unroll
  for (int r = 0; r < 8; ++r) {
    esel[r] = __shfl(ediag, r + 8 * hi, 32);   // e_ii for this lane's row
    inv[r]  = fast_rcp(sacc[r] - esel[r]);     // 1 / (S_i - e_ii)
  }
#pragma unroll
  for (int t = 0; t < 16; ++t) {
    const int col = t * 16 + ln;
#pragma unroll
    for (int r = 0; r < 8; ++r) {
      const int row = i0 + r + 8 * hi;
      float fv = (float)fbf[(size_t)row * DFEAT + col];
      float v  = (acc[t][r] - esel[r] * fv) * inv[r];
      tailbf[(size_t)row * DFEAT + col] = (__bf16)v;
    }
  }
}

// ------------- output GEMM: out = tanh([f, tail] @ Bc^T + bias) -------------
__global__ void __launch_bounds__(128)
k_out_gemm(const __bf16* __restrict__ fbf, const __bf16* __restrict__ tailbf,
           const __bf16* __restrict__ Bc, const float* __restrict__ b_out,
           const float* __restrict__ b_dist, float* __restrict__ out) {
  const int tid  = threadIdx.x;
  const int lane = tid & 31;
  const int wave = tid >> 5;
  const int ln   = lane & 15;
  const int hi   = lane >> 4;
  const int i0   = (blockIdx.x * 4 + wave) * 16;

  v8f acc[16];
#pragma unroll
  for (int t = 0; t < 16; ++t) acc[t] = zero_v8f();

#pragma unroll
  for (int ks = 0; ks < 16; ++ks) {   // K = 512 total: 8 chunks f, 8 chunks tail
    const __bf16* srcA = (ks < 8) ? fbf : tailbf;
    const int k0 = (ks & 7) * 32;
    const __bf16* pa = srcA + (size_t)(i0 + ln) * DFEAT + k0 + hi * 8;
    v16bf va = make_frag(pa, pa + 16);
#pragma unroll
    for (int t = 0; t < 16; ++t) {
      const __bf16* pb = Bc + (size_t)(t * 16 + ln) * 512 + ks * 32 + hi * 8;
      v16bf vb = make_frag(pb, pb + 16);
      acc[t] = __builtin_amdgcn_wmma_f32_16x16x32_bf16(false, va, false, vb,
                                                       (short)0, acc[t], false, false);
    }
  }

  const float CT = 2.8853900817779268f;  // 2 * log2(e), for tanh via exp2
#pragma unroll
  for (int t = 0; t < 16; ++t) {
    const int o = t * 16 + ln;
    const float bias = b_out[o] - b_dist[o];
#pragma unroll
    for (int r = 0; r < 8; ++r) {
      const int row = i0 + r + 8 * hi;
      float x = acc[t][r] + bias;
      x = fminf(fmaxf(x, -15.0f), 15.0f);
      float e2 = fast_exp2(x * CT);
      out[(size_t)row * NOUT + o] = (e2 - 1.0f) * fast_rcp(e2 + 1.0f);
    }
  }
}

// ------------------------------- launcher -------------------------------
extern "C" void kernel_launch(void* const* d_in, const int* in_sizes, int n_in,
                              void* d_out, int out_size, void* d_ws, size_t ws_size,
                              hipStream_t stream) {
  const float* features = (const float*)d_in[0];
  // d_in[1] (key_features) is unused by the reference.
  const float* adj = (const float*)d_in[2];
  const float* Wd  = (const float*)d_in[3];
  const float* bd  = (const float*)d_in[4];
  const float* Wo  = (const float*)d_in[5];
  const float* bo  = (const float*)d_in[6];
  float* out = (float*)d_out;

  char* ws = (char*)d_ws;
  const size_t FB = (size_t)NROW * DFEAT * sizeof(__bf16);  // 4 MB
  __bf16* fbf    = (__bf16*)(ws);            // [N, D] bf16
  __bf16* fT     = (__bf16*)(ws + FB);       // [D, N] bf16 (transposed)
  __bf16* tailbf = (__bf16*)(ws + 2 * FB);   // [N, D] bf16
  __bf16* Bc     = (__bf16*)(ws + 3 * FB);   // [256, 512] bf16 combined weight

  k_prep_feats<<<(NROW * DFEAT + 255) / 256, 256, 0, stream>>>(features, fbf, fT);
  k_prep_w<<<(256 * 512 + 255) / 256, 256, 0, stream>>>(Wd, Wo, Bc);

  // 512 row-strips of 16 rows; 4 waves (strips) per 128-thread workgroup.
  k_softmax_spmm<<<NROW / 64, 128, 0, stream>>>(adj, fT, fbf, tailbf);
  k_out_gemm<<<NROW / 64, 128, 0, stream>>>(fbf, tailbf, Bc, bo, bd, out);
}
